// FaceClassifier_3624952398794
// MI455X (gfx1250) — matrix operand order repaired and checked by reading, not compiled, not run on hardware
//
#include <hip/hip_runtime.h>
#include <hip/hip_bf16.h>

// ---------------- problem constants ----------------
#define NB    4
#define NPER  8192
#define NN    (NB * NPER)      // 32768
#define KNN   20
#define DH    128
#define NEDGE (2 * NN * KNN)   // 1310720  (multiple of 64)

#define EDGE_PER_WAVE  16
#define WAVES_PER_BLK  4
#define EDGE_PER_BLK   (EDGE_PER_WAVE * WAVES_PER_BLK)   // 64

typedef __attribute__((ext_vector_type(16))) __bf16 v16bf;
typedef __attribute__((ext_vector_type(8)))  __bf16 v8bf;
typedef __attribute__((ext_vector_type(8)))  float  v8f;
typedef __attribute__((ext_vector_type(4)))  int    v4i;

__device__ __forceinline__ unsigned short f2bf(float f) {
  // round-to-nearest-even float -> bf16 bits
  unsigned u = __float_as_uint(f);
  unsigned r = u + 0x7FFFu + ((u >> 16) & 1u);
  return (unsigned short)(r >> 16);
}
__device__ __forceinline__ float bf2f(unsigned short h) {
  return __uint_as_float(((unsigned)h) << 16);
}

// ---- async global->LDS copy (CDNA5 ASYNC engine), with sync fallback ----
#if __has_builtin(__builtin_amdgcn_global_load_async_to_lds_b128)
#define HAS_ASYNC_LDS 1
#else
#define HAS_ASYNC_LDS 0
#endif

__device__ __forceinline__ void copy16_g2l(const void* g, void* l) {
#if HAS_ASYNC_LDS
  __builtin_amdgcn_global_load_async_to_lds_b128(
      (__attribute__((address_space(1))) v4i*)(g),
      (__attribute__((address_space(3))) v4i*)(l), 0, 0);
#else
  *(uint4*)l = *(const uint4*)g;
#endif
}
__device__ __forceinline__ void wait_async_lds() {
#if HAS_ASYNC_LDS
#if __has_builtin(__builtin_amdgcn_s_wait_asynccnt)
  __builtin_amdgcn_s_wait_asynccnt(0);
#else
  asm volatile("s_wait_asynccnt 0x0" ::: "memory");
#endif
#endif
}

// ---------------- KNN: per-node top-20 within its graph ----------------
__global__ void __launch_bounds__(256)
knn_kernel(const float* __restrict__ pos, int* __restrict__ knn_idx)
{
  __shared__ float sp[256 * 3];
  const int i    = blockIdx.x * 256 + threadIdx.x;   // node id (grid = NN/256)
  const int base = (i >> 13) << 13;                  // graph start (8192-aligned)
  const float px = pos[i * 3 + 0];
  const float py = pos[i * 3 + 1];
  const float pz = pos[i * 3 + 2];

  float bd[KNN];
  int   bi[KNN];
#pragma unroll
  for (int k = 0; k < KNN; ++k) { bd[k] = 3.4e38f; bi[k] = 0; }

  for (int t = 0; t < NPER; t += 256) {
    for (int idx = threadIdx.x; idx < 256 * 3; idx += 256)
      sp[idx] = pos[(size_t)(base + t) * 3 + idx];
    __syncthreads();
    for (int c = 0; c < 256; ++c) {
      const int j = base + t + c;
      const float dx = px - sp[c * 3 + 0];
      const float dy = py - sp[c * 3 + 1];
      const float dz = pz - sp[c * 3 + 2];
      const float d2 = dx * dx + dy * dy + dz * dz;
      if (j != i && d2 < bd[KNN - 1]) {
        float cd = d2; int ci = j;            // bubble-insert, static indices only
#pragma unroll
        for (int p = 0; p < KNN; ++p) {
          if (cd < bd[p]) {
            float td = bd[p]; int ti = bi[p];
            bd[p] = cd; bi[p] = ci;
            cd = td; ci = ti;
          }
        }
      }
    }
    __syncthreads();
  }
#pragma unroll
  for (int k = 0; k < KNN; ++k) knn_idx[(size_t)i * KNN + k] = bi[k];
}

// ---------------- edge list + dedup weights ----------------
// first half:  e         : src=a, dst=knn(a)[k], w=1
// second half: NN*KNN+e  : src=j, dst=a,         w = (a in knn(j)) ? 0 : 1
__global__ void __launch_bounds__(256)
edge_build_kernel(const int* __restrict__ knn,
                  int* __restrict__ src, int* __restrict__ dst,
                  float* __restrict__ wt)
{
  const int e = blockIdx.x * 256 + threadIdx.x;
  if (e >= NN * KNN) return;
  const int a = e / KNN;
  const int j = knn[e];
  src[e] = a; dst[e] = j; wt[e] = 1.f;
  bool dup = false;
#pragma unroll
  for (int t = 0; t < KNN; ++t) dup = dup || (knn[(size_t)j * KNN + t] == a);
  const int e2 = NN * KNN + e;
  src[e2] = j; dst[e2] = a; wt[e2] = dup ? 0.f : 1.f;
}

// ---------------- one-time weight prep: bf16, fragment-ordered ----------------
// fragment index for B-matrix element (k, n) within a 128x128 tile:
//   kt=k/32, nt=n/16, lane=(n%16)|(((k%32)/16)<<4), elem=k%16
__device__ __forceinline__ int frag_index(int k, int n) {
  const int kt = k >> 5, nt = n >> 4;
  const int fl = (n & 15) | (((k >> 4) & 1) << 4);
  return ((((kt << 3) + nt) << 5) + fl) * 16 + (k & 15);
}

__global__ void __launch_bounds__(256)
prep_weights_kernel(const float* __restrict__ W1s, const float* __restrict__ W2s,
                    unsigned short* __restrict__ w1f, unsigned short* __restrict__ w2f)
{
  const int idx = blockIdx.x * 256 + threadIdx.x;
  if (idx >= 3 * 128 * 128) return;
  const int layer = idx / (128 * 128);
  const int r = idx - layer * 128 * 128;
  const int k = r >> 7, n = r & 127;
  const int d = layer * 16384 + frag_index(k, n);
  w1f[d] = f2bf(W1s[(size_t)layer * 131 * 128 + k * 128 + n]);
  w2f[d] = f2bf(W2s[(size_t)layer * 128 * 128 + k * 128 + n]);
}

__global__ void __launch_bounds__(256)
x_to_bf16_kernel(const float* __restrict__ x, unsigned short* __restrict__ xbf)
{
  const int i = blockIdx.x * 256 + threadIdx.x;   // grid = NN*DH/256
  xbf[i] = f2bf(x[i]);
}

// ---------------- fused edge MLP (the WMMA kernel) ----------------
// per wave: 16 edges -> relu([x_src|rel] @ W1 + b1) @ W2 + b2 -> atomic scatter at dst
__global__ void __launch_bounds__(128)
edge_mlp_kernel(const unsigned short* __restrict__ xbf,   // N x 128 bf16
                const float* __restrict__ pos,
                const int* __restrict__ src, const int* __restrict__ dst,
                const float* __restrict__ wmask,
                const unsigned short* __restrict__ w1f,   // 16384 u16, fragment order
                const unsigned short* __restrict__ w2f,   // 16384 u16, fragment order
                const float* __restrict__ W1,             // 131x128 f32 (rel rows 128..130)
                const float* __restrict__ b1,
                const float* __restrict__ b2,
                float* __restrict__ yout)
{
  __shared__ __align__(32) unsigned short fragbuf[128 * 128];           // 32 KB (W1 then W2)
  __shared__ float s_w1b[3][128];
  __shared__ float s_b1[128];
  __shared__ float s_b2[128];
  __shared__ __align__(32) unsigned short s_x[WAVES_PER_BLK][16 * 128]; // A / hidden staging
  __shared__ float s_rel[WAVES_PER_BLK][16][4];
  __shared__ int   s_dst[WAVES_PER_BLK][16];
  __shared__ float s_wt [WAVES_PER_BLK][16];

  const int tid  = threadIdx.x;
  const int lane = tid & 31;
  const int wv   = tid >> 5;
  const int half = lane >> 4;
  const int nlo  = lane & 15;
  const int e0   = blockIdx.x * EDGE_PER_BLK + wv * EDGE_PER_WAVE;

  // ---- async DMA: W1 fragments (whole block) ----
  for (int c = tid; c < 2048; c += 128)               // 2048 x 16B = 32 KB
    copy16_g2l(w1f + c * 8, &fragbuf[c * 8]);

  // ---- async DMA: 16-row bf16 x gather (per wave) ----
#pragma unroll
  for (int it = 0; it < 8; ++it) {
    const int t   = lane + it * 32;   // 0..255
    const int row = t >> 4;           // 0..15
    const int ch  = t & 15;           // 16B chunk within row
    const int s   = src[e0 + row];
    copy16_g2l(xbf + (size_t)s * 128 + ch * 8, &s_x[wv][row * 128 + ch * 8]);
  }

  // ---- small sync staging: rel weights, biases, per-edge metadata ----
  for (int idx = tid; idx < 3 * 128; idx += 128)
    s_w1b[idx >> 7][idx & 127] = W1[(128 + (idx >> 7)) * 128 + (idx & 127)];
  if (tid < 128) { s_b1[tid] = b1[tid]; s_b2[tid] = b2[tid]; }
  if (lane < 16) {
    const int e = e0 + lane;
    const int s = src[e], d = dst[e];
    s_rel[wv][lane][0] = pos[d * 3 + 0] - pos[s * 3 + 0];
    s_rel[wv][lane][1] = pos[d * 3 + 1] - pos[s * 3 + 1];
    s_rel[wv][lane][2] = pos[d * 3 + 2] - pos[s * 3 + 2];
    s_dst[wv][lane] = d;
    s_wt [wv][lane] = wmask[e];
  }
  wait_async_lds();
  __syncthreads();

  // ---- GEMM1: C init = b1 + rel . W1[128:131]  (folds rel-part + bias) ----
  v8f acc[8];
#pragma unroll
  for (int nt = 0; nt < 8; ++nt) {
    const int n = (nt << 4) + nlo;
    const float wb0 = s_w1b[0][n], wb1 = s_w1b[1][n], wb2 = s_w1b[2][n];
    const float bb = s_b1[n];
#pragma unroll
    for (int r = 0; r < 8; ++r) {
      const int row = r + (half << 3);
      acc[nt][r] = bb + s_rel[wv][row][0] * wb0
                      + s_rel[wv][row][1] * wb1
                      + s_rel[wv][row][2] * wb2;
    }
  }
  const int m = nlo; // A-matrix row for this lane
#pragma unroll
  for (int kt = 0; kt < 4; ++kt) {
    union { v16bf v; v8bf h[2]; } a;
    const unsigned short* ap = &s_x[wv][m * 128 + (kt << 5) + (half << 4)];
    a.h[0] = *(const v8bf*)(ap);        // K = kt*32 + 16*half + [0..7]
    a.h[1] = *(const v8bf*)(ap + 8);    // K = kt*32 + 16*half + 8 + [0..7]
#pragma unroll
    for (int nt = 0; nt < 8; ++nt) {
      const v16bf b = *(const v16bf*)&fragbuf[((((kt << 3) + nt) << 5) + lane) * 16];
      acc[nt] = __builtin_amdgcn_wmma_f32_16x16x32_bf16(
          false, a.v, false, b, (short)0, acc[nt], false, false);
    }
  }
  // ---- ReLU + restage hidden (bf16) into the same per-wave tile ----
#pragma unroll
  for (int nt = 0; nt < 8; ++nt) {
#pragma unroll
    for (int r = 0; r < 8; ++r) {
      float h = acc[nt][r];
      h = h > 0.f ? h : 0.f;
      s_x[wv][(r + (half << 3)) * 128 + (nt << 4) + nlo] = f2bf(h);
    }
  }
  __syncthreads();                       // all waves done reading W1 fragments
  // ---- async DMA: restage fragment buffer with W2 ----
  for (int c = tid; c < 2048; c += 128)
    copy16_g2l(w2f + c * 8, &fragbuf[c * 8]);
  wait_async_lds();
  __syncthreads();

  // ---- GEMM2: C init = b2 ----
  v8f acc2[8];
#pragma unroll
  for (int nt = 0; nt < 8; ++nt) {
    const float bb = s_b2[(nt << 4) + nlo];
#pragma unroll
    for (int r = 0; r < 8; ++r) acc2[nt][r] = bb;
  }
#pragma unroll
  for (int kt = 0; kt < 4; ++kt) {
    union { v16bf v; v8bf h[2]; } a;
    const unsigned short* ap = &s_x[wv][m * 128 + (kt << 5) + (half << 4)];
    a.h[0] = *(const v8bf*)(ap);
    a.h[1] = *(const v8bf*)(ap + 8);
#pragma unroll
    for (int nt = 0; nt < 8; ++nt) {
      const v16bf b = *(const v16bf*)&fragbuf[((((kt << 3) + nt) << 5) + lane) * 16];
      acc2[nt] = __builtin_amdgcn_wmma_f32_16x16x32_bf16(
          false, a.v, false, b, (short)0, acc2[nt], false, false);
    }
  }

  // ---- masked scatter-add at dst (L2-resident accumulator) ----
#pragma unroll
  for (int r = 0; r < 8; ++r) {
    const int row = r + (half << 3);
    const float w = s_wt[wv][row];
    if (w != 0.f) {
      float* yrow = yout + (size_t)s_dst[wv][row] * 128 + nlo;
#pragma unroll
      for (int nt = 0; nt < 8; ++nt)
        __hip_atomic_fetch_add(yrow + (nt << 4), acc2[nt][r],
                               __ATOMIC_RELAXED, __HIP_MEMORY_SCOPE_AGENT);
    }
  }
}

// ---------------- elementwise helpers ----------------
__global__ void zero_kernel(float* __restrict__ p, int n) {
  const int i = blockIdx.x * 256 + threadIdx.x;
  if (i < n) p[i] = 0.f;
}
// relu(y) -> bf16 activation mirror for the next layer
__global__ void relu_bf16_kernel(const float* __restrict__ y,
                                 unsigned short* __restrict__ xbf, int n) {
  const int i = blockIdx.x * 256 + threadIdx.x;
  if (i < n) { const float v = y[i]; xbf[i] = f2bf(v > 0.f ? v : 0.f); }
}

// ---------------- final head: logits + per-graph softmax ----------------
__global__ void __launch_bounds__(256)
logits_kernel(const unsigned short* __restrict__ xbf, const float* __restrict__ Wf,
              const float* __restrict__ bfv, float* __restrict__ logits)
{
  __shared__ float swf[128];
  if (threadIdx.x < 128) swf[threadIdx.x] = Wf[threadIdx.x];
  __syncthreads();
  const int i = blockIdx.x * 256 + threadIdx.x;   // grid = NN/256
  float acc = bfv[0];
#pragma unroll 4
  for (int c = 0; c < 128; c += 4) {
    const ushort4 v = *(const ushort4*)(xbf + (size_t)i * 128 + c);
    acc += bf2f(v.x) * swf[c]     + bf2f(v.y) * swf[c + 1]
         + bf2f(v.z) * swf[c + 2] + bf2f(v.w) * swf[c + 3];
  }
  logits[i] = acc;
}

__global__ void __launch_bounds__(1024)
softmax_kernel(const float* __restrict__ logits, float* __restrict__ out)
{
  __shared__ float red[1024];
  const int tid = threadIdx.x;
  const float* lg = logits + (size_t)blockIdx.x * NPER;
  float* ob = out + (size_t)blockIdx.x * NPER;

  float mx = -3.4e38f;
  for (int i = tid; i < NPER; i += 1024) mx = fmaxf(mx, lg[i]);
  red[tid] = mx; __syncthreads();
  for (int s = 512; s > 0; s >>= 1) {
    if (tid < s) red[tid] = fmaxf(red[tid], red[tid + s]);
    __syncthreads();
  }
  const float m = red[0]; __syncthreads();

  float sm = 0.f;
  for (int i = tid; i < NPER; i += 1024) sm += expf(lg[i] - m);
  red[tid] = sm; __syncthreads();
  for (int s = 512; s > 0; s >>= 1) {
    if (tid < s) red[tid] += red[tid + s];
    __syncthreads();
  }
  const float inv = 1.f / red[0];
  for (int i = tid; i < NPER; i += 1024) ob[i] = expf(lg[i] - m) * inv;
}

// ---------------- driver ----------------
extern "C" void kernel_launch(void* const* d_in, const int* in_sizes, int n_in,
                              void* d_out, int out_size, void* d_ws, size_t ws_size,
                              hipStream_t stream)
{
  const float* x   = (const float*)d_in[0];
  const float* pos = (const float*)d_in[1];
  // d_in[2] = batch (unused: equal-size sorted graphs)
  const float* W1s = (const float*)d_in[3];
  const float* b1s = (const float*)d_in[4];
  const float* W2s = (const float*)d_in[5];
  const float* b2s = (const float*)d_in[6];
  const float* Wf  = (const float*)d_in[7];
  const float* bfv = (const float*)d_in[8];

  char* ws = (char*)d_ws;
  int*   knn    = (int*)ws;   ws += (size_t)NN * KNN * sizeof(int);
  int*   src    = (int*)ws;   ws += (size_t)NEDGE * sizeof(int);
  int*   dst    = (int*)ws;   ws += (size_t)NEDGE * sizeof(int);
  float* wt     = (float*)ws; ws += (size_t)NEDGE * sizeof(float);
  float* y      = (float*)ws; ws += (size_t)NN * DH * sizeof(float);
  unsigned short* xbf0 = (unsigned short*)ws; ws += (size_t)NN * DH * sizeof(unsigned short);
  unsigned short* xbf1 = (unsigned short*)ws; ws += (size_t)NN * DH * sizeof(unsigned short);
  unsigned short* w1f  = (unsigned short*)ws; ws += (size_t)3 * 128 * 128 * sizeof(unsigned short);
  unsigned short* w2f  = (unsigned short*)ws; ws += (size_t)3 * 128 * 128 * sizeof(unsigned short);
  float* logits = (float*)ws;

  knn_kernel<<<NN / 256, 256, 0, stream>>>(pos, knn);
  edge_build_kernel<<<(NN * KNN + 255) / 256, 256, 0, stream>>>(knn, src, dst, wt);
  prep_weights_kernel<<<(3 * 128 * 128 + 255) / 256, 256, 0, stream>>>(W1s, W2s, w1f, w2f);
  x_to_bf16_kernel<<<(NN * DH) / 256, 256, 0, stream>>>(x, xbf0);

  const unsigned short* cur = xbf0;
  unsigned short* nxt[3] = { xbf1, xbf0, xbf1 };
  for (int i = 0; i < 3; ++i) {
    zero_kernel<<<(NN * DH) / 256, 256, 0, stream>>>(y, NN * DH);
    edge_mlp_kernel<<<NEDGE / EDGE_PER_BLK, 128, 0, stream>>>(
        cur, pos, src, dst, wt,
        w1f + (size_t)i * 16384, w2f + (size_t)i * 16384,
        W1s + (size_t)i * 131 * 128, b1s + (size_t)i * 128, b2s + (size_t)i * 128,
        y);
    relu_bf16_kernel<<<(NN * DH) / 256, 256, 0, stream>>>(y, nxt[i], NN * DH);
    cur = nxt[i];
  }

  logits_kernel<<<NN / 256, 256, 0, stream>>>(cur, Wf, bfv, logits);
  softmax_kernel<<<NB, 1024, 0, stream>>>(logits, (float*)d_out);
}
